// RnnLM_38070590112433
// MI455X (gfx1250) — matrix operand order repaired and checked by reading, not compiled, not run on hardware
//
#include <hip/hip_runtime.h>
#include <hip/hip_bf16.h>
#include <hip/hip_fp16.h>

// Problem dims (fixed by the reference)
#define BB 8
#define TT 512
#define LL 4
#define DD 1024
#define VV 32000
#define KDIM 1024                // contraction dim for both WMMA GEMMs

typedef __attribute__((ext_vector_type(16))) _Float16 v16h;
typedef __attribute__((ext_vector_type(8)))  _Float16 v8h;
typedef __attribute__((ext_vector_type(8)))  float    v8f;

// LDS tile geometry: 128Mx32K (A) + 128Nx32K (B), rows padded 32->40 halves
// (80B row stride => b128 LDS reads tile all 64 banks conflict-free).
#define ROW_HALVES   40          // 32 data halves + 8 pad halves
#define ROW_BYTES    80
#define A_REGION_B   (128 * ROW_BYTES)          // 10240 bytes
#define BUF_BYTES    (2 * A_REGION_B)           // 20480 bytes (A + B regions)
#define BUF_HALVES   (BUF_BYTES / 2)            // 10240 halves

// ---------------------------------------------------------------------------
// Utility kernels
// ---------------------------------------------------------------------------
__global__ void zero_f32_kernel(float* __restrict__ p, int n) {
    int i = blockIdx.x * blockDim.x + threadIdx.x;
    int stride = gridDim.x * blockDim.x;
    for (; i < n; i += stride) p[i] = 0.0f;
}

__global__ void cvt_f32_to_f16_kernel(const float* __restrict__ src,
                                      _Float16* __restrict__ dst, int n) {
    int i = blockIdx.x * blockDim.x + threadIdx.x;
    int stride = gridDim.x * blockDim.x;
    for (; i < n; i += stride) dst[i] = (_Float16)src[i];
}

// Embedding gather: emb[row, :] = wte[ids[row], :], row = b*T + t
__global__ void embed_kernel(const int* __restrict__ ids,
                             const float* __restrict__ wte,
                             float* __restrict__ emb) {
    int row = blockIdx.x;                      // 0..B*T-1
    int id  = ids[row];
    const float4* src = (const float4*)(wte + (size_t)id * DD);
    float4*       dst = (float4*)(emb + (size_t)row * DD);
    dst[threadIdx.x] = src[threadIdx.x];       // 256 threads * float4 = 1024 f32
}

// ---------------------------------------------------------------------------
// One recurrence step for one layer (serial over t via stream ordering).
// h_new[b][j] = tanh( whh[j,:]·hprev[b,:] + whhb[j] + wxh[j,:]·x[b,:] + wxhb[j] )
// 32 blocks x 256 threads: j = blk*32 + (tid>>3), b = tid&7.
// All recurrent weights (32 MiB) stay L2-resident (192 MB L2).
// ---------------------------------------------------------------------------
__global__ void rnn_step_kernel(const float* __restrict__ whh,
                                const float* __restrict__ whhb,
                                const float* __restrict__ wxh,
                                const float* __restrict__ wxhb,
                                const float* __restrict__ hprev,  // [B, D]
                                const float* __restrict__ x,      // base, stride xstride per b
                                long xstride,
                                float* __restrict__ hout,         // [B, D]
                                float* __restrict__ h3out,        // optional, stride h3stride per b
                                long h3stride) {
    const int j = blockIdx.x * 32 + (threadIdx.x >> 3);
    const int b = threadIdx.x & 7;

    const float4* wr  = (const float4*)(whh + (size_t)j * DD);
    const float4* hr  = (const float4*)(hprev + (size_t)b * DD);
    const float4* wr2 = (const float4*)(wxh + (size_t)j * DD);
    const float4* xv  = (const float4*)(x + (size_t)b * xstride);

    float acc = 0.0f;
#pragma unroll 4
    for (int k = 0; k < DD / 4; ++k) {
        float4 w = wr[k], h = hr[k];
        acc += w.x * h.x + w.y * h.y + w.z * h.z + w.w * h.w;
    }
#pragma unroll 4
    for (int k = 0; k < DD / 4; ++k) {
        float4 w = wr2[k], v = xv[k];
        acc += w.x * v.x + w.y * v.y + w.z * v.z + w.w * v.w;
    }
    acc += whhb[j] + wxhb[j];
    float hn = tanhf(acc);
    hout[(size_t)b * DD + j] = hn;
    if (h3out) h3out[(size_t)b * h3stride + j] = hn;
}

// ---------------------------------------------------------------------------
// LDS-tiled WMMA NT GEMM: C[m,n] = sum_k A[m,k]*B[n,k] (+ bias[n]), K = 1024.
// A:[M,K] f16, B:[N,K] f16 (row-major, K contiguous), C:[M,N] f32.
// Block = 256 threads = 8 waves; block tile 128M x 128N, k-step 32,
// double-buffered LDS staging via global_load_async_to_lds_b128 (ASYNCcnt).
// Wave w owns rows [16w,16w+16) x 128N: one A fragment feeds 8 WMMAs, and
// every staged B byte is consumed by all 8 waves.
// M%128==0, N%128==0 at all call sites -> EXEC is all-ones at every WMMA.
// ---------------------------------------------------------------------------
__global__ void __launch_bounds__(256)
wmma_gemm_nt_lds_kernel(const _Float16* __restrict__ A,
                        const _Float16* __restrict__ Bm,
                        const float* __restrict__ bias,
                        float* __restrict__ C,
                        int M, int N) {
    __shared__ __align__(16) _Float16 smem[2 * BUF_HALVES];   // 40 KB

    const int tid  = threadIdx.x;
    const int lane = tid & 31;
    const int wave = tid >> 5;
    const int gn0  = blockIdx.x * 128;
    const int gm0  = blockIdx.y * 128;

    // Escape the LDS base address into a memory-clobbering asm once: the
    // async-to-LDS writes below are then potential stores through this
    // pointer, so the ds_load fragments cannot be folded to undef.
    asm volatile("" :: "v"(&smem[0]) : "memory");

    // Per-thread staging plan: 4 x 16B chunks per thread per tile, only the
    // k offset changes across iterations. chunk c = tid + 256*i:
    //   c <  512 -> A region, row c>>2, 16B part c&3
    //   c >= 512 -> B region likewise.
    const _Float16* stage_src[4];   // row base (k=0) in global
    unsigned        stage_lds[4];   // byte offset inside one LDS buffer
#pragma unroll
    for (int i = 0; i < 4; ++i) {
        const int c = tid + 256 * i;             // uniform region per (wave,i)
        if (c < 512) {
            const int row = c >> 2, part = c & 3;
            stage_src[i] = A + (size_t)(gm0 + row) * KDIM + part * 8;
            stage_lds[i] = (unsigned)(row * ROW_BYTES + part * 16);
        } else {
            const int c2 = c - 512;
            const int row = c2 >> 2, part = c2 & 3;
            stage_src[i] = Bm + (size_t)(gn0 + row) * KDIM + part * 8;
            stage_lds[i] = (unsigned)A_REGION_B + (unsigned)(row * ROW_BYTES + part * 16);
        }
    }

    auto stage = [&](int kk, unsigned bufBaseBytes) {
#pragma unroll
        for (int i = 0; i < 4; ++i) {
            const _Float16* src = stage_src[i] + kk;
            const unsigned ldsOff = bufBaseBytes + stage_lds[i];
            asm volatile("global_load_async_to_lds_b128 %0, %1, off"
                         :: "v"(ldsOff), "v"(src)
                         : "memory");
        }
    };

    v8f acc[8] = {};

    const int mrow  = wave * 16 + (lane & 15);
    const int khalf = lane >> 4;

    // B fragment (32x16): lane l -> n=l&15; halves0-7 K=(l>>4)*8+i, halves8-15 K=16+(l>>4)*8+i
    auto loadB = [&](const _Float16* Bs, int j) -> v16h {
        const int nrow = j * 16 + (lane & 15);
        const _Float16* bp = Bs + nrow * ROW_HALVES + khalf * 8;
        v8h b_lo = *(const v8h*)bp;
        v8h b_hi = *(const v8h*)(bp + 16);
        v16h b;
#pragma unroll
        for (int i = 0; i < 8; ++i) { b[i] = b_lo[i]; b[i + 8] = b_hi[i]; }
        return b;
    };

    auto compute = [&](const _Float16* As) {
        const _Float16* Bs = As + (A_REGION_B / 2);

        // A fragment (16x32): lane l -> m=l&15; halves0-7 K=khalf*8+i,
        // halves8-15 K=16+khalf*8+i  (two conflict-free ds_load_b128)
        v8h a_lo = *(const v8h*)(As + mrow * ROW_HALVES + khalf * 8);
        v8h a_hi = *(const v8h*)(As + mrow * ROW_HALVES + 16 + khalf * 8);
        v16h a;
#pragma unroll
        for (int i = 0; i < 8; ++i) { a[i] = a_lo[i]; a[i + 8] = a_hi[i]; }

        // 2-stage B pipeline: fragment j+1 is in flight while WMMA consumes j
        // (only 2 B fragments live -> no register-pressure shuffling).
        v16h bc = loadB(Bs, 0);
#pragma unroll
        for (int j = 0; j < 7; ++j) {
            v16h bn = loadB(Bs, j + 1);
            acc[j] = __builtin_amdgcn_wmma_f32_16x16x32_f16(
                false, a, false, bc, (short)0, acc[j], false, false);
            bc = bn;
        }
        acc[7] = __builtin_amdgcn_wmma_f32_16x16x32_f16(
            false, a, false, bc, (short)0, acc[7], false, false);
    };

    constexpr int ksteps = KDIM >> 5;            // 32

    unsigned curBase = 0u, nxtBase = (unsigned)BUF_BYTES;
    stage(0, curBase);

    // Main loop (peeled tail): always stage next tile, wait only for current
    // (async loads complete in order; the 4 just-issued stay outstanding).
    // unroll(disable): one loop body copy keeps acc pinned in one reg range.
#pragma clang loop unroll(disable)
    for (int it = 0; it < ksteps - 1; ++it) {
        stage((it + 1) << 5, nxtBase);
        asm volatile("s_wait_asynccnt 0x4" ::: "memory");
        __syncthreads();
        compute(smem + (curBase >> 1));
        __syncthreads();   // everyone done reading before buffer is re-staged
        unsigned t = curBase; curBase = nxtBase; nxtBase = t;
    }
    asm volatile("s_wait_asynccnt 0x0" ::: "memory");
    __syncthreads();
    compute(smem + (curBase >> 1));

    // Writeback: C v8f layout -> elem r: M = r + 8*(lane>>4), N = lane&15
#pragma unroll
    for (int j = 0; j < 8; ++j) {
        const int ncol = gn0 + j * 16 + (lane & 15);
        const float bv = bias ? bias[ncol] : 0.0f;
#pragma unroll
        for (int r = 0; r < 8; ++r) {
            const int m = gm0 + wave * 16 + (lane >> 4) * 8 + r;
            C[(size_t)m * N + ncol] = acc[j][r] + bv;
        }
    }
}

// ---------------------------------------------------------------------------
// Host orchestration
// ---------------------------------------------------------------------------
extern "C" void kernel_launch(void* const* d_in, const int* in_sizes, int n_in,
                              void* d_out, int out_size, void* d_ws, size_t ws_size,
                              hipStream_t stream) {
    const int*   ids   = (const int*)  d_in[0];
    const float* wte   = (const float*)d_in[1];
    const float* wxh_w = (const float*)d_in[2];
    const float* wxh_b = (const float*)d_in[3];
    const float* whh_w = (const float*)d_in[4];
    const float* whh_b = (const float*)d_in[5];
    const float* who_w = (const float*)d_in[6];
    const float* who_b = (const float*)d_in[7];
    float*       out   = (float*)d_out;          // [B,T,V] == [B*T, V] row-major

    char*  ws  = (char*)d_ws;
    size_t off = 0;
    auto alloc = [&](size_t bytes) -> char* {
        off = (off + 255) & ~(size_t)255;
        char* p = ws + off;
        off += bytes;
        return p;
    };

    const size_t BT = (size_t)BB * TT;           // 4096
    float*    hA    = (float*)   alloc(sizeof(float) * LL * BB * DD);
    float*    hBuf  = (float*)   alloc(sizeof(float) * LL * BB * DD);
    float*    emb   = (float*)   alloc(sizeof(float) * BT * DD);
    float*    h3    = (float*)   alloc(sizeof(float) * BT * DD);
    float*    proj  = (float*)   alloc(sizeof(float) * BT * DD);
    _Float16* h3h   = (_Float16*)alloc(sizeof(_Float16) * BT * DD);
    _Float16* projh = (_Float16*)alloc(sizeof(_Float16) * BT * DD);
    _Float16* wteh  = (_Float16*)alloc(sizeof(_Float16) * (size_t)VV * DD);
    _Float16* whoh  = (_Float16*)alloc(sizeof(_Float16) * (size_t)DD * DD);

    // 1) zero recurrent state every call (ws is not re-poisoned by harness)
    zero_f32_kernel<<<64, 256, 0, stream>>>(hA,   LL * BB * DD);
    zero_f32_kernel<<<64, 256, 0, stream>>>(hBuf, LL * BB * DD);

    // 2) embedding gather
    embed_kernel<<<(int)BT, 256, 0, stream>>>(ids, wte, emb);

    // 3) weight down-converts for the WMMA GEMMs
    cvt_f32_to_f16_kernel<<<4096, 256, 0, stream>>>(wte, wteh, VV * DD);
    cvt_f32_to_f16_kernel<<<1024, 256, 0, stream>>>(who_w, whoh, DD * DD);

    // 4) serial recurrence: 512 steps x 4 layers, double-buffered hidden state
    for (int t = 0; t < TT; ++t) {
        float* prev = (t & 1) ? hBuf : hA;
        float* cur  = (t & 1) ? hA   : hBuf;
        for (int l = 0; l < LL; ++l) {
            const float* xbase;
            long xstride;
            if (l == 0) { xbase = emb + (size_t)t * DD; xstride = (long)TT * DD; }
            else        { xbase = cur + (size_t)(l - 1) * BB * DD; xstride = DD; }
            float* h3out = (l == LL - 1) ? (h3 + (size_t)t * DD) : nullptr;
            rnn_step_kernel<<<32, 256, 0, stream>>>(
                whh_w + (size_t)l * DD * DD, whh_b + (size_t)l * DD,
                wxh_w + (size_t)l * DD * DD, wxh_b + (size_t)l * DD,
                prev + (size_t)l * BB * DD, xbase, xstride,
                cur + (size_t)l * BB * DD, h3out, (long)TT * DD);
        }
    }

    // 5) output projection: proj = h3 @ who^T + who_b   [4096 x 1024]
    cvt_f32_to_f16_kernel<<<2048, 256, 0, stream>>>(h3, h3h, (int)(BT * DD));
    {
        dim3 grid(DD / 128, (int)(BT / 128));
        wmma_gemm_nt_lds_kernel<<<grid, 256, 0, stream>>>(h3h, whoh, who_b, proj,
                                                          (int)BT, DD);
    }

    // 6) tied logits: out = proj @ wte^T   [4096 x 32000]  (dominant FLOPs)
    cvt_f32_to_f16_kernel<<<2048, 256, 0, stream>>>(proj, projh, (int)(BT * DD));
    {
        dim3 grid(VV / 128, (int)(BT / 128));
        wmma_gemm_nt_lds_kernel<<<grid, 256, 0, stream>>>(projh, wteh, nullptr, out,
                                                          (int)BT, VV);
    }
}
